// Model_83519934038704
// MI455X (gfx1250) — matrix operand order repaired and checked by reading, not compiled, not run on hardware
//
#include <hip/hip_runtime.h>
#include <math.h>

typedef __attribute__((ext_vector_type(16))) _Float16 v16h;
typedef __attribute__((ext_vector_type(8)))  float    v8f;

#define NN    100000
#define EE    800000
#define IND   128
#define HIDD  64
#define OUTD  40
#define NITER 25   // 20 fixed-point solve + 5 phantom-gradient steps (identical ops)

// ---------- helpers: convert 8 contiguous f32 -> f16 (two float4 loads) ----------
__device__ __forceinline__ void load8(const float* __restrict__ p, _Float16* d) {
  const float4* q = (const float4*)p;
  float4 a = q[0], b = q[1];
  d[0] = (_Float16)a.x; d[1] = (_Float16)a.y; d[2] = (_Float16)a.z; d[3] = (_Float16)a.w;
  d[4] = (_Float16)b.x; d[5] = (_Float16)b.y; d[6] = (_Float16)b.z; d[7] = (_Float16)b.w;
}
__device__ __forceinline__ void load8_relu(const float* __restrict__ p, _Float16* d) {
  const float4* q = (const float4*)p;
  float4 a = q[0], b = q[1];
  d[0] = (_Float16)fmaxf(a.x, 0.f); d[1] = (_Float16)fmaxf(a.y, 0.f);
  d[2] = (_Float16)fmaxf(a.z, 0.f); d[3] = (_Float16)fmaxf(a.w, 0.f);
  d[4] = (_Float16)fmaxf(b.x, 0.f); d[5] = (_Float16)fmaxf(b.y, 0.f);
  d[6] = (_Float16)fmaxf(b.z, 0.f); d[7] = (_Float16)fmaxf(b.w, 0.f);
}

// ================= fused encoder + bias-module GEMMs (WMMA) =================
// h = x @ W_enc^T + b_enc   (per-wave 16x128 * 128x64 tile)
// b = h @ W_bias^T          (h staged via LDS for C->A re-layout)
__global__ __launch_bounds__(256)
void enc_bias_kernel(const float* __restrict__ x, const float* __restrict__ W_enc,
                     const float* __restrict__ b_enc, const float* __restrict__ W_bias,
                     float* __restrict__ bvec) {
  __shared__ float lds_h[8][16][68];           // stride 68: bank-conflict-free, 16B aligned
  const int wave = threadIdx.x >> 5;
  const int lane = threadIdx.x & 31;
  const int n    = lane & 15;                  // tile column (and A-row) index
  const int g    = lane >> 4;                  // lane-group
  const int base = blockIdx.x * 128 + wave * 16;
  int nodec = base + n; if (nodec >= NN) nodec = NN - 1;   // clamp loads, guard stores

  union { _Float16 h[16]; v16h v; } A, B;
  v8f acc[4];
#pragma unroll
  for (int jt = 0; jt < 4; ++jt) {
    float bi = b_enc[jt * 16 + n];
#pragma unroll
    for (int j = 0; j < 8; ++j) acc[jt][j] = bi;           // fold b_enc into C init
  }

  const float* xrow = x + (size_t)nodec * IND;
#pragma unroll
  for (int c = 0; c < 4; ++c) {                            // K = 128 in chunks of 32
    load8(xrow + 32 * c + 8 * g,      A.h);                // A: k = 32c+8g+i
    load8(xrow + 32 * c + 8 * g + 16, A.h + 8);            //    k = 32c+8g+16+i
#pragma unroll
    for (int jt = 0; jt < 4; ++jt) {
      const float* wr = W_enc + (size_t)(jt * 16 + n) * IND + 32 * c + 8 * g;
      load8(wr,     B.h);                                  // B: k = 32c+8g+i
      load8(wr + 16, B.h + 8);
      acc[jt] = __builtin_amdgcn_wmma_f32_16x16x32_f16(
          false, A.v, false, B.v, (short)0, acc[jt], false, false);
    }
  }

  // spill h tile (C layout: m = j+8g, col = 16jt+n) to LDS
#pragma unroll
  for (int jt = 0; jt < 4; ++jt)
#pragma unroll
    for (int j = 0; j < 8; ++j)
      lds_h[wave][j + 8 * g][jt * 16 + n] = acc[jt][j];
  __syncthreads();

  v8f acc2[4];
#pragma unroll
  for (int jt = 0; jt < 4; ++jt)
#pragma unroll
    for (int j = 0; j < 8; ++j) acc2[jt][j] = 0.f;

  const float* hrow = &lds_h[wave][n][0];
#pragma unroll
  for (int c = 0; c < 2; ++c) {                            // K = 64 in chunks of 32
    load8(hrow + 32 * c + 8 * g,      A.h);
    load8(hrow + 32 * c + 8 * g + 16, A.h + 8);
#pragma unroll
    for (int jt = 0; jt < 4; ++jt) {
      const float* wr = W_bias + (size_t)(jt * 16 + n) * HIDD + 32 * c + 8 * g;
      load8(wr,     B.h);
      load8(wr + 16, B.h + 8);
      acc2[jt] = __builtin_amdgcn_wmma_f32_16x16x32_f16(
          false, A.v, false, B.v, (short)0, acc2[jt], false, false);
    }
  }

#pragma unroll
  for (int j = 0; j < 8; ++j) {
    int row = base + j + 8 * g;
    if (row < NN) {
#pragma unroll
      for (int jt = 0; jt < 4; ++jt)
        bvec[(size_t)row * HIDD + jt * 16 + n] = acc2[jt][j];
    }
  }
}

// ======================= CSR construction =======================
__global__ __launch_bounds__(256)
void hist_kernel(const int* __restrict__ dst, int* __restrict__ cnt) {
  for (int e = blockIdx.x * blockDim.x + threadIdx.x; e < EE; e += gridDim.x * blockDim.x)
    atomicAdd(&cnt[dst[e]], 1);
}

__global__ __launch_bounds__(1024)
void scan_kernel(int* __restrict__ cursor, int* __restrict__ rowptr) {
  __shared__ int sd[1024];
  __shared__ int carry;
  const int tid = threadIdx.x;
  if (tid == 0) carry = 0;
  __syncthreads();
  for (int basei = 0; basei < NN; basei += 1024) {
    int i = basei + tid;
    int v = (i < NN) ? cursor[i] : 0;
    sd[tid] = v;
    __syncthreads();
    for (int off = 1; off < 1024; off <<= 1) {
      int t = (tid >= off) ? sd[tid - off] : 0;
      __syncthreads();
      sd[tid] += t;
      __syncthreads();
    }
    int excl = carry + sd[tid] - v;
    if (i < NN) { rowptr[i] = excl; cursor[i] = excl; }   // cursor doubles as fill cursors
    __syncthreads();
    if (tid == 0) carry += sd[1023];
    __syncthreads();
  }
  if (tid == 0) rowptr[NN] = carry;
}

__global__ __launch_bounds__(256)
void fill_kernel(const int* __restrict__ src, const int* __restrict__ dst,
                 const float* __restrict__ w, int* __restrict__ cursor,
                 int* __restrict__ col, float* __restrict__ wgt) {
  for (int e = blockIdx.x * blockDim.x + threadIdx.x; e < EE; e += gridDim.x * blockDim.x) {
    int p = atomicAdd(&cursor[dst[e]], 1);
    col[p] = src[e];
    wgt[p] = w[e];
  }
}

// =============== one fixed-point / phantom-gradient step (L2-resident) ===============
// un = gamma*relu(beta*(P u) + b) + (1-gamma)*u ; 64 threads per node, k = lane
__global__ __launch_bounds__(256)
void prop_kernel(const float* __restrict__ u, const float* __restrict__ b,
                 const int* __restrict__ rowptr, const int* __restrict__ col,
                 const float* __restrict__ wgt, float* __restrict__ un,
                 const float* __restrict__ beta_p, const float* __restrict__ gamma_p) {
  int node = blockIdx.x * 4 + (threadIdx.x >> 6);
  int k    = threadIdx.x & 63;
  if (node >= NN) return;
  float beta  = 1.f / (1.f + __expf(-beta_p[0]));
  float gamma = 1.f / (1.f + __expf(-gamma_p[0]));
  int s = rowptr[node], e = rowptr[node + 1];
  float agg = 0.f;
  for (int p = s; p < e; ++p)
    agg = fmaf(wgt[p], u[(size_t)col[p] * HIDD + k], agg);   // 256B contiguous per edge
  float uo = u[(size_t)node * HIDD + k];
  float v  = fmaf(beta, agg, b[(size_t)node * HIDD + k]);
  v = v > 0.f ? v : 0.f;
  un[(size_t)node * HIDD + k] = fmaf(gamma, v - uo, uo);
}

// ======================= decoder: out = relu(u) @ W_dec^T (WMMA) =======================
__global__ __launch_bounds__(256)
void dec_kernel(const float* __restrict__ u, const float* __restrict__ W_dec,
                float* __restrict__ out) {
  const int wave = threadIdx.x >> 5;
  const int lane = threadIdx.x & 31;
  const int n    = lane & 15;
  const int g    = lane >> 4;
  const int base = blockIdx.x * 128 + wave * 16;
  int nodec = base + n; if (nodec >= NN) nodec = NN - 1;

  union { _Float16 h[16]; v16h v; } A, B;
  v8f acc[3];
#pragma unroll
  for (int jt = 0; jt < 3; ++jt)
#pragma unroll
    for (int j = 0; j < 8; ++j) acc[jt][j] = 0.f;

  const float* urow = u + (size_t)nodec * HIDD;
#pragma unroll
  for (int c = 0; c < 2; ++c) {
    load8_relu(urow + 32 * c + 8 * g,      A.h);
    load8_relu(urow + 32 * c + 8 * g + 16, A.h + 8);
#pragma unroll
    for (int jt = 0; jt < 3; ++jt) {
      int j = jt * 16 + n;
      if (j < OUTD) {
        const float* wr = W_dec + (size_t)j * HIDD + 32 * c + 8 * g;
        load8(wr,     B.h);
        load8(wr + 16, B.h + 8);
      } else {
#pragma unroll
        for (int i = 0; i < 16; ++i) B.h[i] = (_Float16)0.f;
      }
      acc[jt] = __builtin_amdgcn_wmma_f32_16x16x32_f16(
          false, A.v, false, B.v, (short)0, acc[jt], false, false);
    }
  }

#pragma unroll
  for (int j = 0; j < 8; ++j) {
    int row = base + j + 8 * g;
    if (row < NN) {
#pragma unroll
      for (int jt = 0; jt < 3; ++jt) {
        int cidx = jt * 16 + n;
        if (cidx < OUTD) out[(size_t)row * OUTD + cidx] = acc[jt][j];
      }
    }
  }
}

// ======================= launcher =======================
extern "C" void kernel_launch(void* const* d_in, const int* in_sizes, int n_in,
                              void* d_out, int out_size, void* d_ws, size_t ws_size,
                              hipStream_t stream) {
  (void)in_sizes; (void)n_in; (void)out_size; (void)ws_size;
  const float* x       = (const float*)d_in[0];
  const int*   ei      = (const int*)d_in[1];     // [2,E]: src = ei, dst = ei+E
  const float* ew      = (const float*)d_in[2];
  const float* W_enc   = (const float*)d_in[3];
  const float* b_enc   = (const float*)d_in[4];
  const float* W_bias  = (const float*)d_in[5];
  const float* W_dec   = (const float*)d_in[6];
  const float* beta_p  = (const float*)d_in[7];
  const float* gamma_p = (const float*)d_in[8];
  float* out = (float*)d_out;

  char* ws = (char*)d_ws;
  size_t off = 0;
  auto carve = [&](size_t bytes) -> void* {
    void* p = ws + off;
    off = (off + bytes + 255) & ~(size_t)255;
    return p;
  };
  float* bvec   = (float*)carve((size_t)NN * HIDD * 4);
  float* u0     = (float*)carve((size_t)NN * HIDD * 4);
  float* u1     = (float*)carve((size_t)NN * HIDD * 4);
  int*   rowptr = (int*)carve((size_t)(NN + 1) * 4);
  int*   cursor = (int*)carve((size_t)NN * 4);
  int*   col    = (int*)carve((size_t)EE * 4);
  float* wgt    = (float*)carve((size_t)EE * 4);

  hipMemsetAsync(cursor, 0, (size_t)NN * 4, stream);
  hipMemsetAsync(u0, 0, (size_t)NN * HIDD * 4, stream);        // u0 = zeros_like(b)

  hist_kernel<<<1024, 256, 0, stream>>>(ei + EE, cursor);
  scan_kernel<<<1, 1024, 0, stream>>>(cursor, rowptr);
  fill_kernel<<<1024, 256, 0, stream>>>(ei, ei + EE, ew, cursor, col, wgt);

  enc_bias_kernel<<<(NN + 127) / 128, 256, 0, stream>>>(x, W_enc, b_enc, W_bias, bvec);

  float* uc = u0;
  float* un = u1;
  for (int it = 0; it < NITER; ++it) {
    prop_kernel<<<(NN + 3) / 4, 256, 0, stream>>>(uc, bvec, rowptr, col, wgt, un,
                                                  beta_p, gamma_p);
    float* t = uc; uc = un; un = t;
  }

  dec_kernel<<<(NN + 127) / 128, 256, 0, stream>>>(uc, W_dec, out);
}
